// SocialPooling_66322884985171
// MI455X (gfx1250) — matrix operand order repaired and checked by reading, not compiled, not run on hardware
//
#include <hip/hip_runtime.h>
#include <hip/hip_bf16.h>

typedef __attribute__((ext_vector_type(16))) _Float16 v16h;
typedef __attribute__((ext_vector_type(8)))  float    v8f;

#define AREA_HALF 0.8f
#define EPS_F     0.01f
#define CELL_F    0.8f

// ---------------------------------------------------------------------------
// Prepass: repack ht (N x 32, f32, row-major) into f16 B-fragments laid out in
// this code's contiguous-k 32x16 B operand order, so the main loop loads one v16h
// (32B, coalesced) per fragment.
//   bpack[(((t*2 + f)*32 + L)*16) + e] = (f16) ht[(t*32 + Kb)*32 + f*16 + (L&15)]
//   with Kb = (L>>4)*16 + e   (B layout: lane owns N=L&15, element e holds K)
// ---------------------------------------------------------------------------
__global__ void SocialPooling_pack_b(const float* __restrict__ ht,
                                     _Float16* __restrict__ bpack, int total) {
    int idx = blockIdx.x * blockDim.x + threadIdx.x;
    if (idx >= total) return;
    int e = idx & 15;
    int L = (idx >> 4) & 31;
    int f = (idx >> 9) & 1;
    int t = idx >> 10;
    int kb = ((L >> 4) << 4) | e;
    int j  = t * 32 + kb;
    int n  = f * 16 + (L & 15);
    bpack[idx] = (_Float16)ht[(size_t)j * 32 + n];
}

// ---------------------------------------------------------------------------
// Main kernel: 4 waves / block, one 16-row i-tile per wave.
// For each 32-wide j-tile inside this tile's scene range:
//   build 4 binary mask A-fragments (one per grid cell), load 2 B-fragments
//   (feature columns 0-15 / 16-31), issue 8 x v_wmma_f32_16x16x32_f16.
//
// Grid-cell index: for valid pairs, floor((r+0.8f)/0.8f) under RN f32 is
// bit-identical to ((r + 0.8f) >= 0.8f)  (threshold of the rounded divide is
// exactly 0.8f), so no divide/floor is needed.
// ---------------------------------------------------------------------------
template <bool PACKED>
__global__ void SocialPooling_wmma(const float* __restrict__ ht,
                                   const float* __restrict__ pos,
                                   const int*   __restrict__ scene,
                                   const _Float16* __restrict__ bpack,
                                   float* __restrict__ out, int N) {
    const int lane  = threadIdx.x & 31;
    const int wave  = threadIdx.x >> 5;
    const int itile = blockIdx.x * 4 + wave;
    const int i0    = itile * 16;
    if (i0 >= N) return;

    // This lane's i-row for the mask (A layout: M = lane & 15 in both halves)
    const int   m   = lane & 15;
    const int   i   = i0 + m;
    const float pix = pos[i * 2 + 0];
    const float piy = pos[i * 2 + 1];
    const int   si  = scene[i];

    // Scene ids are sorted: binary-search the contiguous valid j range.
    const int s_lo = scene[i0];
    const int s_hi = scene[i0 + 15];
    int lo = 0, hi = N;
    while (lo < hi) { int mid = (lo + hi) >> 1; if (scene[mid] <  s_lo) lo = mid + 1; else hi = mid; }
    const int jlo = lo;
    lo = 0; hi = N;
    while (lo < hi) { int mid = (lo + hi) >> 1; if (scene[mid] <= s_hi) lo = mid + 1; else hi = mid; }
    const int jhi = lo;
    const int t0 = jlo >> 5;
    const int t1 = (jhi + 31) >> 5;

    const float LIM = AREA_HALF - EPS_F;   // |r| < LIM  ==  two-sided ref check
    const _Float16 oneh  = (_Float16)1.0f;
    const _Float16 zeroh = (_Float16)0.0f;

    v8f acc[4][2] = {};

    for (int t = t0; t < t1; ++t) {
        const int j0 = t * 32;

        if (PACKED) {
            // prefetch next tile's packed B fragments into cache
            if (t + 1 < t1)
                __builtin_prefetch(bpack + ((size_t)(t + 1) * 2 * 32 + lane) * 16, 0, 3);
        }

        // lane l holds metadata of j = j0 + l
        const int   jme = j0 + lane;
        const float pjx = pos[jme * 2 + 0];
        const float pjy = pos[jme * 2 + 1];
        const int   sj  = scene[jme];

        // Build the four mask A-fragments.
        v16h a[4];
#pragma unroll
        for (int e = 0; e < 16; ++e) {
            // A layout: element e of lane L holds K = (L>>4)*16 + e
            const int srcK = ((lane >> 4) << 4) | e;
            const float qx = __shfl(pjx, srcK, 32);
            const float qy = __shfl(pjy, srcK, 32);
            const int   qs = __shfl(sj,  srcK, 32);
            const int   j  = j0 + srcK;

            const float rx = qx - pix;
            const float ry = qy - piy;
            // within-box (abs-modifier compare), same scene, not self
            const bool inb = (__builtin_fabsf(rx) < LIM) &&
                             (__builtin_fabsf(ry) < LIM) &&
                             (qs == si) && (j != i);
            // grid cell bits: exact match of floor((r+0.8f)/0.8f) on valid range
            const bool gx = (rx + AREA_HALF) >= CELL_F;
            const bool gy = (ry + AREA_HALF) >= CELL_F;

            a[0][e] = (inb && !gx && !gy) ? oneh : zeroh;
            a[1][e] = (inb && !gx &&  gy) ? oneh : zeroh;
            a[2][e] = (inb &&  gx && !gy) ? oneh : zeroh;
            a[3][e] = (inb &&  gx &&  gy) ? oneh : zeroh;
        }

        // Load the two B fragments (feature columns 0-15 and 16-31).
        v16h b0, b1;
        if (PACKED) {
            b0 = *(const v16h*)(bpack + (((size_t)t * 2 + 0) * 32 + lane) * 16);
            b1 = *(const v16h*)(bpack + (((size_t)t * 2 + 1) * 32 + lane) * 16);
        } else {
#pragma unroll
            for (int e = 0; e < 16; ++e) {
                const int kb = ((lane >> 4) << 4) | e;
                const float* hrow = ht + (size_t)(j0 + kb) * 32 + (lane & 15);
                b0[e] = (_Float16)hrow[0];
                b1[e] = (_Float16)hrow[16];
            }
        }

        // 8 WMMAs: 4 grid cells x 2 halves of H.
#pragma unroll
        for (int g = 0; g < 4; ++g) {
            acc[g][0] = __builtin_amdgcn_wmma_f32_16x16x32_f16(
                false, a[g], false, b0, (short)0, acc[g][0], false, false);
            acc[g][1] = __builtin_amdgcn_wmma_f32_16x16x32_f16(
                false, a[g], false, b1, (short)0, acc[g][1], false, false);
        }
    }

    // Store D: element r of lane L is (M = r + 8*(L>>4), N = L&15).
    const int nn    = lane & 15;
    const int mbase = (lane >> 4) * 8;
#pragma unroll
    for (int g = 0; g < 4; ++g)
#pragma unroll
        for (int f = 0; f < 2; ++f)
#pragma unroll
            for (int r = 0; r < 8; ++r) {
                const int irow = i0 + mbase + r;
                out[((size_t)irow * 4 + g) * 32 + f * 16 + nn] = acc[g][f][r];
            }
}

extern "C" void kernel_launch(void* const* d_in, const int* in_sizes, int n_in,
                              void* d_out, int out_size, void* d_ws, size_t ws_size,
                              hipStream_t stream) {
    const float* ht    = (const float*)d_in[0];
    const float* pos   = (const float*)d_in[1];
    const int*   scene = (const int*)d_in[2];
    float*       out   = (float*)d_out;
    const int N = in_sizes[2];      // 8192
    const int H = 32;

    const size_t packBytes = (size_t)N * H * 2;   // f16 repack of ht
    const int blocks = (N + 63) / 64;             // 4 waves/block, 16 rows/wave

    if (ws_size >= packBytes) {
        _Float16* bpack = (_Float16*)d_ws;
        const int total = N * H;
        SocialPooling_pack_b<<<(total + 255) / 256, 256, 0, stream>>>(ht, bpack, total);
        SocialPooling_wmma<true><<<blocks, 128, 0, stream>>>(ht, pos, scene, bpack, out, N);
    } else {
        SocialPooling_wmma<false><<<blocks, 128, 0, stream>>>(ht, pos, scene, nullptr, out, N);
    }
}
